// attention_net_18717467476006
// MI455X (gfx1250) — matrix operand order repaired and checked by reading, not compiled, not run on hardware
//
#include <hip/hip_runtime.h>
#include <hip/hip_bf16.h>
#include <math.h>

typedef __attribute__((ext_vector_type(16))) _Float16 v16h;
typedef __attribute__((ext_vector_type(8)))  _Float16 v8h;
typedef __attribute__((ext_vector_type(8)))  float    v8f;

#define PADW   112
#define PS     112
#define TOPN   4
#define NANCH  1614
#define NVOC   1001
#define NPAD   1008

// ---------------------------------------------------------------------------
// Prep: f32 -> f16 copy (backbone conv weights, keeps [512][768] layout)
// ---------------------------------------------------------------------------
__global__ void cvt_f16_kernel(const float* __restrict__ in, _Float16* __restrict__ out,
                               int n) {
  const int i = blockIdx.x * blockDim.x + threadIdx.x;
  if (i < n) out[i] = (_Float16)in[i];
}

// Prep: FC weight [K][N] f32 -> transposed, zero-padded [Npad][K] f16
__global__ void fcw_t_f16_kernel(const float* __restrict__ W, _Float16* __restrict__ out,
                                 int K, int N, int Npad) {
  const int i = blockIdx.x * blockDim.x + threadIdx.x;
  if (i >= Npad * K) return;
  const int n = i / K, k = i - n * K;
  out[i] = (n < N) ? (_Float16)W[(size_t)k * N + n] : (_Float16)0.f;
}

// ---------------------------------------------------------------------------
// Patchify-conv GEMM + bias + ReLU + fused GAP accumulate.
// A: implicit 16x16/stride-16 patches of src [B,3,IH,IH] -> M = B*S, K = 768
// B: pre-converted f16 weights [512][768].  N = 512.
// One wave32 computes one 16-row M-tile x 8 N-tiles: A fragment built once,
// then 8 back-to-back v_wmma_f32_16x16x32_f16 per K-step (24 K-steps).
// uniformTile!=0 => whole M-tile lies in one image: in-register row reduction
// + cross-half shuffle => 1 atomic per output column (16x fewer atomics).
// ---------------------------------------------------------------------------
__global__ void patch_gemm_gap(const float* __restrict__ src,
                               const _Float16* __restrict__ w16,  // [512][768] f16
                               const float* __restrict__ bias,    // [512]
                               float* __restrict__ acc,           // [B,512]
                               int B, int IH, int W16, int S, int uniformTile) {
  const int wave = (blockIdx.x * blockDim.x + threadIdx.x) >> 5;
  const int lane = threadIdx.x & 31;
  const int Mtiles = (B * S) >> 4;
  const int total  = Mtiles * 4;           // 4 groups of 8 N-tiles (N=512)
  if (wave >= total) return;               // uniform per wave
  const int mtile = wave >> 2;
  const int ngrp  = wave & 3;
  const int hi = lane >> 4;
  const int lm = lane & 15;

  // A row for this lane (always in-bounds: B*S is a multiple of 16)
  const int gm = mtile * 16 + lm;
  const int b  = gm / S;
  const int p  = gm - b * S;
  const int oy = p / W16;
  const int ox = p - oy * W16;
  const float* img = src + (size_t)b * 3 * IH * IH;
  const int y0 = oy * 16, x0 = ox * 16;

  // this lane's base weight row: columns n_t = ngrp*128 + t*16 + lm
  const _Float16* wbase = w16 + (size_t)(ngrp * 128 + lm) * 768;

  v8f accv[8] = {};
  for (int kb = 0; kb < 24; ++kb) {
    const int kbase = kb * 32;
    // --- A fragment: elems 0..7 -> K = kbase+hi*8+j ; 8..15 -> +16 more ---
    const int k0 = kbase + hi * 8;
    const int c0 = k0 >> 8, r0 = k0 & 255;
    const float* pa0 = img + ((size_t)c0 * IH + (y0 + (r0 >> 4))) * IH + x0 + (r0 & 15);
    const int k1 = kbase + 16 + hi * 8;
    const int c1 = k1 >> 8, r1 = k1 & 255;
    const float* pa1 = img + ((size_t)c1 * IH + (y0 + (r1 >> 4))) * IH + x0 + (r1 & 15);
    float4 a0 = *reinterpret_cast<const float4*>(pa0);
    float4 a1 = *reinterpret_cast<const float4*>(pa0 + 4);
    float4 a2 = *reinterpret_cast<const float4*>(pa1);
    float4 a3 = *reinterpret_cast<const float4*>(pa1 + 4);
    v16h av;
    av[0]=(_Float16)a0.x; av[1]=(_Float16)a0.y; av[2]=(_Float16)a0.z; av[3]=(_Float16)a0.w;
    av[4]=(_Float16)a1.x; av[5]=(_Float16)a1.y; av[6]=(_Float16)a1.z; av[7]=(_Float16)a1.w;
    av[8]=(_Float16)a2.x; av[9]=(_Float16)a2.y; av[10]=(_Float16)a2.z; av[11]=(_Float16)a2.w;
    av[12]=(_Float16)a3.x; av[13]=(_Float16)a3.y; av[14]=(_Float16)a3.z; av[15]=(_Float16)a3.w;
    // --- 8 N-tiles: B fragment = 2x8 f16 at K = kbase+hi*8 and +16, no conversion ---
    #pragma unroll
    for (int t = 0; t < 8; ++t) {
      const _Float16* pb = wbase + (size_t)t * 16 * 768 + kbase + hi * 8;
      const v8h blo = *reinterpret_cast<const v8h*>(pb);
      const v8h bhi = *reinterpret_cast<const v8h*>(pb + 16);
      v16h bv;
      #pragma unroll
      for (int e = 0; e < 8; ++e) { bv[e] = blo[e]; bv[8 + e] = bhi[e]; }
      accv[t] = __builtin_amdgcn_wmma_f32_16x16x32_f16(false, av, false, bv,
                                                       (short)0, accv[t], false, false);
    }
  }
  // bias + ReLU + GAP accumulate. C layout: lane L, reg r -> M=r+8*hi, N=L&15
  if (uniformTile) {
    const int b2 = (mtile * 16) / S;       // whole tile in one image
    #pragma unroll
    for (int t = 0; t < 8; ++t) {
      const int n = ngrp * 128 + t * 16 + lm;
      const float bs = bias[n];
      float s = 0.f;
      #pragma unroll
      for (int r = 0; r < 8; ++r) {
        float v = accv[t][r] + bs;
        s += (v > 0.f ? v : 0.f);
      }
      s += __shfl_xor(s, 16, 32);          // combine M halves (same column n)
      if (hi == 0) atomicAdd(&acc[(size_t)b2 * 512 + n], s);
    }
  } else {
    #pragma unroll
    for (int t = 0; t < 8; ++t) {
      const int n = ngrp * 128 + t * 16 + lm;
      const float bs = bias[n];
      #pragma unroll
      for (int r = 0; r < 8; ++r) {
        const int gm2 = mtile * 16 + r + 8 * hi;
        const int b2  = gm2 / S;
        float v = accv[t][r] + bs;
        v = v > 0.f ? v : 0.f;
        atomicAdd(&acc[(size_t)b2 * 512 + n], v);
      }
    }
  }
}

// ---------------------------------------------------------------------------
// Dense FC GEMM via WMMA with pre-transposed zero-padded f16 weights [Npad][K].
// out[M,N] = A[M,K] @ W + bias.  M mult of 16, K mult of 32, Npad = 63*16.
// ---------------------------------------------------------------------------
__global__ void fc_gemm_wmma(const float* __restrict__ A, const _Float16* __restrict__ WT,
                             const float* __restrict__ bias, float* __restrict__ out,
                             int M, int K, int N) {
  const int wave = (blockIdx.x * blockDim.x + threadIdx.x) >> 5;
  const int lane = threadIdx.x & 31;
  const int Ntiles = (N + 15) >> 4;
  const int total  = (M >> 4) * Ntiles;
  if (wave >= total) return;
  const int mtile = wave / Ntiles;
  const int ntile = wave - mtile * Ntiles;
  const int hi = lane >> 4, lm = lane & 15;
  const int gm = mtile * 16 + lm;
  const int n  = ntile * 16 + lm;          // < Npad always; WT rows zero-padded
  const float* arow = A + (size_t)gm * K;
  const _Float16* wrow = WT + (size_t)n * K;

  v8f c = {};
  for (int kb = 0; kb < K; kb += 32) {
    const float* pa0 = arow + kb + hi * 8;
    const float* pa1 = arow + kb + 16 + hi * 8;
    float4 a0 = *reinterpret_cast<const float4*>(pa0);
    float4 a1 = *reinterpret_cast<const float4*>(pa0 + 4);
    float4 a2 = *reinterpret_cast<const float4*>(pa1);
    float4 a3 = *reinterpret_cast<const float4*>(pa1 + 4);
    v16h av;
    av[0]=(_Float16)a0.x; av[1]=(_Float16)a0.y; av[2]=(_Float16)a0.z; av[3]=(_Float16)a0.w;
    av[4]=(_Float16)a1.x; av[5]=(_Float16)a1.y; av[6]=(_Float16)a1.z; av[7]=(_Float16)a1.w;
    av[8]=(_Float16)a2.x; av[9]=(_Float16)a2.y; av[10]=(_Float16)a2.z; av[11]=(_Float16)a2.w;
    av[12]=(_Float16)a3.x; av[13]=(_Float16)a3.y; av[14]=(_Float16)a3.z; av[15]=(_Float16)a3.w;
    const _Float16* pb = wrow + kb + hi * 8;
    const v8h blo = *reinterpret_cast<const v8h*>(pb);
    const v8h bhi = *reinterpret_cast<const v8h*>(pb + 16);
    v16h bv;
    #pragma unroll
    for (int e = 0; e < 8; ++e) { bv[e] = blo[e]; bv[8 + e] = bhi[e]; }
    c = __builtin_amdgcn_wmma_f32_16x16x32_f16(false, av, false, bv, (short)0, c, false, false);
  }
  if (n < N) {
    const float bs = bias[n];
    #pragma unroll
    for (int r = 0; r < 8; ++r) {
      const int gm2 = mtile * 16 + r + 8 * hi;
      out[(size_t)gm2 * N + n] = c[r] + bs;
    }
  }
}

// ---------------------------------------------------------------------------
// Generic 3x3 conv (pad=1) + ReLU — ProposalNet trunk (small, VALU).
// ---------------------------------------------------------------------------
__global__ void conv3x3(const float* __restrict__ in, const float* __restrict__ w,
                        const float* __restrict__ bias, float* __restrict__ out,
                        int B, int Cin, int Hin, int Cout, int Hout, int stride) {
  const int idx = blockIdx.x * blockDim.x + threadIdx.x;
  const int tot = B * Cout * Hout * Hout;
  if (idx >= tot) return;
  int t = idx;
  const int ox = t % Hout; t /= Hout;
  const int oy = t % Hout; t /= Hout;
  const int co = t % Cout; const int b = t / Cout;
  float acc = bias[co];
  for (int ci = 0; ci < Cin; ++ci) {
    const float* ip = in + ((size_t)(b * Cin + ci)) * Hin * Hin;
    const float* wp = w + ((size_t)(co * Cin + ci)) * 9;
    #pragma unroll
    for (int ky = 0; ky < 3; ++ky) {
      const int iy = oy * stride + ky - 1;
      if (iy < 0 || iy >= Hin) continue;
      #pragma unroll
      for (int kx = 0; kx < 3; ++kx) {
        const int ix = ox * stride + kx - 1;
        if (ix < 0 || ix >= Hin) continue;
        acc = fmaf(ip[iy * Hin + ix], wp[ky * 3 + kx], acc);
      }
    }
  }
  out[idx] = acc > 0.f ? acc : 0.f;
}

// 1x1 tidy convs -> concatenated rpn_score [B,1614] (torch .view channel-major)
__global__ void tidy_scores(const float* __restrict__ d1, const float* __restrict__ d2,
                            const float* __restrict__ d3,
                            const float* __restrict__ w1, const float* __restrict__ b1,
                            const float* __restrict__ w2, const float* __restrict__ b2,
                            const float* __restrict__ w3, const float* __restrict__ b3,
                            float* __restrict__ score, int B) {
  const int idx = blockIdx.x * blockDim.x + threadIdx.x;
  if (idx >= B * NANCH) return;
  const int j = idx % NANCH;
  const int b = idx / NANCH;
  float acc;
  if (j < 1176) {                       // t1: 6 ch x 196
    const int ch = j / 196, pos = j - ch * 196;
    acc = b1[ch];
    const float* f = d1 + (size_t)b * 128 * 196;
    for (int c = 0; c < 128; ++c) acc = fmaf(f[c * 196 + pos], w1[ch * 128 + c], acc);
  } else if (j < 1470) {                // t2: 6 ch x 49
    const int jj = j - 1176; const int ch = jj / 49, pos = jj - ch * 49;
    acc = b2[ch];
    const float* f = d2 + (size_t)b * 128 * 49;
    for (int c = 0; c < 128; ++c) acc = fmaf(f[c * 49 + pos], w2[ch * 128 + c], acc);
  } else {                              // t3: 9 ch x 16
    const int jj = j - 1470; const int ch = jj / 16, pos = jj - ch * 16;
    acc = b3[ch];
    const float* f = d3 + (size_t)b * 128 * 16;
    for (int c = 0; c < 128; ++c) acc = fmaf(f[c * 16 + pos], w3[ch * 128 + c], acc);
  }
  score[idx] = acc;
}

// ---------------------------------------------------------------------------
// Greedy hard-NMS, one block per batch. Scores in LDS; argmax matches
// jnp.argmax (lowest index wins ties). Suppresses IoU > 0.25 (incl. self).
// ---------------------------------------------------------------------------
__global__ void nms_kernel(const float* __restrict__ score, const int* __restrict__ anchors,
                           int* __restrict__ topIdx, float* __restrict__ outIdxF,
                           float* __restrict__ outProb) {
  __shared__ float sc[NANCH];
  __shared__ float rv[256];
  __shared__ int   ri[256];
  __shared__ int   ksel;
  const int b = blockIdx.x, tid = threadIdx.x;
  const float* s = score + (size_t)b * NANCH;
  for (int j = tid; j < NANCH; j += 256) sc[j] = s[j];
  __syncthreads();
  for (int it = 0; it < TOPN; ++it) {
    float best = -1e30f; int bi = NANCH;
    for (int j = tid; j < NANCH; j += 256) {
      const float v = sc[j];
      if (v > best || (v == best && j < bi)) { best = v; bi = j; }
    }
    rv[tid] = best; ri[tid] = bi;
    __syncthreads();
    for (int off = 128; off > 0; off >>= 1) {
      if (tid < off) {
        const float v = rv[tid + off]; const int i2 = ri[tid + off];
        if (v > rv[tid] || (v == rv[tid] && i2 < ri[tid])) { rv[tid] = v; ri[tid] = i2; }
      }
      __syncthreads();
    }
    if (tid == 0) {
      ksel = ri[0];
      topIdx[b * TOPN + it]  = ksel;
      outIdxF[b * TOPN + it] = (float)ksel;
      outProb[b * TOPN + it] = s[ksel];
    }
    __syncthreads();
    const int k = ksel;
    const float ky0 = (float)anchors[k*4+0], kx0 = (float)anchors[k*4+1];
    const float ky1 = (float)anchors[k*4+2], kx1 = (float)anchors[k*4+3];
    const float a1 = (ky1 - ky0) * (kx1 - kx0);
    for (int j = tid; j < NANCH; j += 256) {
      const float y0 = (float)anchors[j*4+0], x0 = (float)anchors[j*4+1];
      const float y1 = (float)anchors[j*4+2], x1 = (float)anchors[j*4+3];
      const float inter = fmaxf(fminf(ky1, y1) - fmaxf(ky0, y0), 0.f) *
                          fmaxf(fminf(kx1, x1) - fmaxf(kx0, x0), 0.f);
      const float a2 = (y1 - y0) * (x1 - x0);
      if (inter / (a1 + a2 - inter) > 0.25f) sc[j] = -1e30f;
    }
    __syncthreads();
  }
}

// Bilinear crop-resize (align_corners) from implicitly zero-padded image.
__global__ void crop_kernel(const float* __restrict__ x, const int* __restrict__ anchors,
                            const int* __restrict__ topIdx, float* __restrict__ parts, int B) {
  const int idx = blockIdx.x * blockDim.x + threadIdx.x;
  const int tot = B * TOPN * 3 * PS * PS;
  if (idx >= tot) return;
  int t = idx;
  const int px = t % PS; t /= PS;
  const int py = t % PS; t /= PS;
  const int c  = t % 3;  t /= 3;
  const int tp = t % TOPN; const int b = t / TOPN;
  const int k = topIdx[b * TOPN + tp];
  const float y0 = (float)anchors[k*4+0], x0 = (float)anchors[k*4+1];
  const float y1 = (float)anchors[k*4+2], x1 = (float)anchors[k*4+3];
  const float ys = y0 + (py * (1.f / (PS - 1))) * (y1 - y0 - 1.f);
  const float xs = x0 + (px * (1.f / (PS - 1))) * (x1 - x0 - 1.f);
  const float yf = floorf(ys), xf = floorf(xs);
  const float wy = ys - yf, wx = xs - xf;
  const int Hp = 448 + 2 * PADW;
  const int y0i = min(max((int)yf, 0), Hp - 1);
  const int x0i = min(max((int)xf, 0), Hp - 1);
  const int y1i = min(y0i + 1, Hp - 1);
  const int x1i = min(x0i + 1, Hp - 1);
  auto fetch = [&](int yy, int xx) -> float {
    yy -= PADW; xx -= PADW;
    if (yy < 0 || yy >= 448 || xx < 0 || xx >= 448) return 0.f;
    return x[(((size_t)b * 3 + c) * 448 + yy) * 448 + xx];
  };
  const float top = fetch(y0i, x0i) * (1.f - wx) + fetch(y0i, x1i) * wx;
  const float bot = fetch(y1i, x0i) * (1.f - wx) + fetch(y1i, x1i) * wx;
  parts[idx] = top * (1.f - wy) + bot * wy;
}

__global__ void scale_kernel(const float* __restrict__ a, float* __restrict__ o,
                             float s, int n) {
  const int i = blockIdx.x * blockDim.x + threadIdx.x;
  if (i < n) o[i] = a[i] * s;
}

__global__ void concat_kernel(const float* __restrict__ partFeat,
                              const float* __restrict__ feat,
                              float* __restrict__ out, int B) {
  const int idx = blockIdx.x * blockDim.x + threadIdx.x;
  if (idx >= B * 2560) return;
  const int j = idx % 2560; const int b = idx / 2560;
  out[idx] = (j < 2048)
    ? partFeat[((size_t)b * 4 + (j >> 9)) * 512 + (j & 511)]
    : feat[(size_t)b * 512 + (j - 2048)];
}

extern "C" void kernel_launch(void* const* d_in, const int* in_sizes, int n_in,
                              void* d_out, int out_size, void* d_ws, size_t ws_size,
                              hipStream_t stream) {
  const float* x       = (const float*)d_in[0];
  const float* rpn     = (const float*)d_in[1];
  const int*   anchors = (const int*)  d_in[2];
  const float* p_w1 = (const float*)d_in[3];  const float* p_b1 = (const float*)d_in[4];
  const float* p_w2 = (const float*)d_in[5];  const float* p_b2 = (const float*)d_in[6];
  const float* p_w3 = (const float*)d_in[7];  const float* p_b3 = (const float*)d_in[8];
  const float* p_wt1 = (const float*)d_in[9];  const float* p_bt1 = (const float*)d_in[10];
  const float* p_wt2 = (const float*)d_in[11]; const float* p_bt2 = (const float*)d_in[12];
  const float* p_wt3 = (const float*)d_in[13]; const float* p_bt3 = (const float*)d_in[14];
  const float* bb_cw = (const float*)d_in[15]; const float* bb_cb = (const float*)d_in[16];
  const float* bb_fw = (const float*)d_in[17]; const float* bb_fb = (const float*)d_in[18];
  const float* concat_w = (const float*)d_in[19]; const float* concat_b = (const float*)d_in[20];
  const float* part_w   = (const float*)d_in[21]; const float* part_b   = (const float*)d_in[22];

  float* out = (float*)d_out;
  float* ws  = (float*)d_ws;
  // workspace layout (float-element offsets, all 16B aligned)
  float* featAcc16 = ws + 0;          // 16*512
  float* featAcc64 = ws + 8192;       // 64*512
  float* feature16 = ws + 40960;      // 16*512
  float* partFeat  = ws + 49152;      // 64*512
  float* d1        = ws + 81920;      // 16*128*196
  float* d2        = ws + 483328;     // 16*128*49
  float* d3        = ws + 583680;     // 16*128*16
  float* rpnScore  = ws + 616448;     // 16*1614
  int*   topIdxI   = (int*)(ws + 642272); // 64 ints
  float* partImgs  = ws + 642336;     // 64*3*112*112   (ends at 3050784)
  _Float16* w16      = (_Float16*)(ws + 3050784); // 512*768 f16
  _Float16* fwT16    = (_Float16*)(ws + 3247392); // 1008*512 f16
  _Float16* partT16  = (_Float16*)(ws + 3505440); // 1008*512 f16
  _Float16* concatT16= (_Float16*)(ws + 3763488); // 1008*2560 f16 (ends 5053728 floats)

  // output layout (floats)
  float* rawLogits    = out;                  // 16*1001
  float* concatLogits = out + 16016;          // 16*1001
  float* partLogits   = out + 32032;          // 64*1001
  float* topIdxOut    = out + 96096;          // 16*4
  float* topProbOut   = out + 96160;          // 16*4
  float* concatOut    = out + 96224;          // 16*2560

  hipMemsetAsync(featAcc16, 0, (size_t)(8192 + 32768) * sizeof(float), stream);

  // 0) weight prep: f16 conv weights + transposed/padded f16 FC weights
  cvt_f16_kernel<<<1536, 256, 0, stream>>>(bb_cw, w16, 512 * 768);
  fcw_t_f16_kernel<<<(NPAD * 512 + 255) / 256, 256, 0, stream>>>(bb_fw, fwT16, 512, NVOC, NPAD);
  fcw_t_f16_kernel<<<(NPAD * 512 + 255) / 256, 256, 0, stream>>>(part_w, partT16, 512, NVOC, NPAD);
  fcw_t_f16_kernel<<<(NPAD * 2560 + 255) / 256, 256, 0, stream>>>(concat_w, concatT16, 2560, NVOC, NPAD);

  // 1) backbone on full image: GEMM(12544,768,512) via WMMA + fused GAP
  patch_gemm_gap<<<392, 256, 0, stream>>>(x, w16, bb_cb, featAcc16, 16, 448, 28, 784, 1);
  scale_kernel<<<(8192 + 255) / 256, 256, 0, stream>>>(featAcc16, feature16, 1.f / 784.f, 8192);
  fc_gemm_wmma<<<8, 256, 0, stream>>>(feature16, fwT16, bb_fb, rawLogits, 16, 512, NVOC);

  // 2) proposal net
  conv3x3<<<(401408 + 255) / 256, 256, 0, stream>>>(rpn, p_w1, p_b1, d1, 16, 512, 14, 128, 14, 1);
  conv3x3<<<(100352 + 255) / 256, 256, 0, stream>>>(d1, p_w2, p_b2, d2, 16, 128, 14, 128, 7, 2);
  conv3x3<<<(32768 + 255) / 256, 256, 0, stream>>>(d2, p_w3, p_b3, d3, 16, 128, 7, 128, 4, 2);
  tidy_scores<<<(16 * NANCH + 255) / 256, 256, 0, stream>>>(d1, d2, d3, p_wt1, p_bt1,
                                                            p_wt2, p_bt2, p_wt3, p_bt3,
                                                            rpnScore, 16);
  // 3) NMS + crops
  nms_kernel<<<16, 256, 0, stream>>>(rpnScore, anchors, topIdxI, topIdxOut, topProbOut);
  crop_kernel<<<(2408448 + 255) / 256, 256, 0, stream>>>(x, anchors, topIdxI, partImgs, 16);

  // 4) backbone on parts: GEMM(3136,768,512); S=49 not tile-uniform -> per-row atomics
  patch_gemm_gap<<<98, 256, 0, stream>>>(partImgs, w16, bb_cb, featAcc64, 64, 112, 7, 49, 0);
  scale_kernel<<<(32768 + 255) / 256, 256, 0, stream>>>(featAcc64, partFeat, 1.f / 49.f, 32768);
  fc_gemm_wmma<<<32, 256, 0, stream>>>(partFeat, partT16, part_b, partLogits, 64, 512, NVOC);

  // 5) concat head
  concat_kernel<<<(40960 + 255) / 256, 256, 0, stream>>>(partFeat, feature16, concatOut, 16);
  fc_gemm_wmma<<<8, 256, 0, stream>>>(concatOut, concatT16, concat_b, concatLogits, 16, 2560, NVOC);
}
